// LGRLClassifierDecoder_22058952032962
// MI455X (gfx1250) — matrix operand order repaired and checked by reading, not compiled, not run on hardware
//
#include <hip/hip_runtime.h>

typedef __bf16 bf16;
typedef __attribute__((ext_vector_type(16))) __bf16 v16bf;
typedef __attribute__((ext_vector_type(8)))  float  v8f;

#define T_TOK 65536
#define BSEG  64
#define XDIM  512
#define YDIM  2560
#define HDIM  512

union ABFrag { v16bf v; float4 q[2]; };

__device__ __forceinline__ unsigned fkey(float x) {
  unsigned b = __float_as_uint(x);
  return (b & 0x80000000u) ? ~b : (b | 0x80000000u);
}
__device__ __forceinline__ float fdecode(unsigned k) {
  unsigned b = (k & 0x80000000u) ? (k & 0x7FFFFFFFu) : ~k;
  return __uint_as_float(b);
}

// ---------------- prep: convert / transpose-convert to bf16 ----------------
__global__ __launch_bounds__(256) void conv_kernel(const float* __restrict__ S,
                                                   bf16* __restrict__ D, int n) {
  int i = blockIdx.x * 256 + threadIdx.x;
  if (i < n) D[i] = (bf16)S[i];
}

// S is K x N row-major; D is N x K row-major: D[n*K+k] = S[k*N+n]
__global__ __launch_bounds__(256) void tconv_kernel(const float* __restrict__ S,
                                                    bf16* __restrict__ D, int K, int N) {
  long long i = (long long)blockIdx.x * 256 + threadIdx.x;
  long long total = (long long)K * N;
  if (i >= total) return;
  int k = (int)(i / N), n = (int)(i % N);   // coalesced read of S
  D[(long long)n * K + k] = (bf16)S[i];
}

// ---------------- seg_bias = io_flat(64x2560) @ W1b(2560x512) + b1 ---------
__global__ __launch_bounds__(256) void seg_bias_kernel(
    const bf16* __restrict__ ioB, const bf16* __restrict__ W1bT,
    const float* __restrict__ b1, float* __restrict__ seg_bias) {
  const int wave = threadIdx.x >> 5, lane = threadIdx.x & 31;
  const int tile = blockIdx.x * 8 + wave;      // 0..127 : 4 m-tiles x 32 n-tiles
  const int mt = tile >> 5, nt = tile & 31;
  const int r = lane & 15, hr = lane >> 4;
  const bf16* arow = ioB  + (size_t)(mt * 16 + r) * YDIM;
  const bf16* brow = W1bT + (size_t)(nt * 16 + r) * YDIM;
  v8f acc = {0.f, 0.f, 0.f, 0.f, 0.f, 0.f, 0.f, 0.f};
  for (int k0 = 0; k0 < YDIM; k0 += 32) {
    ABFrag a, b;
    a.q[0] = *(const float4*)(arow + k0 + hr * 8);
    a.q[1] = *(const float4*)(arow + k0 + 16 + hr * 8);
    b.q[0] = *(const float4*)(brow + k0 + hr * 8);
    b.q[1] = *(const float4*)(brow + k0 + 16 + hr * 8);
    acc = __builtin_amdgcn_wmma_f32_16x16x32_bf16(false, a.v, false, b.v,
                                                  (short)0, acc, false, false);
  }
  const int col = nt * 16 + r;
  #pragma unroll
  for (int e = 0; e < 8; ++e) {
    const int row = mt * 16 + e + hr * 8;
    seg_bias[(size_t)row * HDIM + col] = acc[e] + b1[col];
  }
}

// ---------------- fused 3-layer MLP -> logits ------------------------------
__global__ __launch_bounds__(256) void mlp_logits_kernel(
    const float* __restrict__ ps, const int* __restrict__ segids,
    const bf16* __restrict__ W1aT, const bf16* __restrict__ W2T,
    const float* __restrict__ seg_bias, const float* __restrict__ b2,
    const float* __restrict__ W3, const float* __restrict__ b3,
    float* __restrict__ logits) {
  constexpr int MT = 64, LDW = XDIM + 8;      // pad 16B to spread LDS banks
  __shared__ __align__(16) bf16 smem[2][MT][LDW];
  __shared__ float red[MT];
  __shared__ int seg_sh[MT];

  const int tid = threadIdx.x;
  const int m0 = blockIdx.x * MT;

  // stage ps tile (64x512 f32) -> bf16 in LDS buffer 0
  for (int i = tid; i < MT * (XDIM / 4); i += 256) {
    const int row = i / (XDIM / 4);
    const int c = (i % (XDIM / 4)) * 4;
    float4 v = *(const float4*)(ps + (size_t)(m0 + row) * XDIM + c);
    smem[0][row][c + 0] = (bf16)v.x;
    smem[0][row][c + 1] = (bf16)v.y;
    smem[0][row][c + 2] = (bf16)v.z;
    smem[0][row][c + 3] = (bf16)v.w;
  }
  if (tid < MT) { seg_sh[tid] = segids[m0 + tid]; red[tid] = 0.0f; }
  __syncthreads();

  const int wave = tid >> 5, lane = tid & 31;
  const int msub = wave & 3;        // 4 m-subtiles of 16 rows
  const int nhalf = wave >> 2;      // 2 waves split the 512 output cols
  const int r = lane & 15, hr = lane >> 4;
  const int arow = msub * 16 + r;

  auto layer = [&](int sb, int db, const bf16* __restrict__ BTg, int mode) {
    const v8f zero = {0.f, 0.f, 0.f, 0.f, 0.f, 0.f, 0.f, 0.f};
    for (int nq = 0; nq < 2; ++nq) {
      v8f acc[8];
      #pragma unroll
      for (int i = 0; i < 8; ++i) acc[i] = zero;
      const int ntb = nhalf * 16 + nq * 8;
      for (int k0 = 0; k0 < XDIM; k0 += 32) {
        ABFrag a;
        a.q[0] = *(const float4*)(&smem[sb][arow][k0 + hr * 8]);
        a.q[1] = *(const float4*)(&smem[sb][arow][k0 + 16 + hr * 8]);
        #pragma unroll
        for (int i = 0; i < 8; ++i) {
          const bf16* brow = BTg + (size_t)((ntb + i) * 16 + r) * XDIM + k0 + hr * 8;
          ABFrag b;
          b.q[0] = *(const float4*)(brow);
          b.q[1] = *(const float4*)(brow + 16);
          acc[i] = __builtin_amdgcn_wmma_f32_16x16x32_bf16(false, a.v, false, b.v,
                                                           (short)0, acc[i], false, false);
        }
      }
      #pragma unroll
      for (int i = 0; i < 8; ++i) {
        const int col = (ntb + i) * 16 + r;
        #pragma unroll
        for (int e = 0; e < 8; ++e) {
          const int row = msub * 16 + e + hr * 8;
          float v = acc[i][e];
          v += (mode == 0) ? seg_bias[(size_t)seg_sh[row] * HDIM + col] : b2[col];
          v = v > 0.0f ? v : 0.0f;
          smem[db][row][col] = (bf16)v;
        }
      }
    }
    __syncthreads();
  };

  layer(0, 1, W1aT, 0);   // h1 = relu(ps@W1a + seg_bias[seg])
  layer(1, 0, W2T, 1);    // h2 = relu(h1@W2 + b2)

  // logits = h2 . W3 + b3   (4 threads per row, LDS-atomic reduce)
  const int row3 = tid & 63, q3 = tid >> 6;
  float p = 0.0f;
  #pragma unroll 4
  for (int j = q3 * 128; j < q3 * 128 + 128; ++j)
    p += (float)smem[0][row3][j] * W3[j];
  atomicAdd(&red[row3], p);
  __syncthreads();
  if (tid < MT) logits[m0 + tid] = red[tid] + b3[0];
}

// ---------------- segment softmax + pooling --------------------------------
__global__ __launch_bounds__(256) void init_kernel(float* pooled, unsigned* segmax_key,
                                                   float* segsum, unsigned* seg_lo,
                                                   unsigned* seg_hi) {
  int i = blockIdx.x * 256 + threadIdx.x;
  if (i < BSEG * XDIM) pooled[i] = 0.0f;
  if (i < BSEG) { segmax_key[i] = 0u; segsum[i] = 0.0f; seg_lo[i] = T_TOK; seg_hi[i] = 0u; }
}

__global__ __launch_bounds__(256) void seg_stats_kernel(const float* __restrict__ logits,
                                                        const int* __restrict__ seg,
                                                        unsigned* segmax_key,
                                                        unsigned* seg_lo, unsigned* seg_hi) {
  int t = blockIdx.x * 256 + threadIdx.x;
  if (t >= T_TOK) return;
  int s = seg[t];
  atomicMax(&segmax_key[s], fkey(logits[t]));
  atomicMin(&seg_lo[s], (unsigned)t);
  atomicMax(&seg_hi[s], (unsigned)(t + 1));
}

__global__ __launch_bounds__(256) void exp_sum_kernel(const float* __restrict__ logits,
                                                      const int* __restrict__ seg,
                                                      const unsigned* __restrict__ segmax_key,
                                                      float* __restrict__ wtmp, float* segsum) {
  int t = blockIdx.x * 256 + threadIdx.x;
  if (t >= T_TOK) return;
  int s = seg[t];
  float e = __expf(logits[t] - fdecode(segmax_key[s]));
  wtmp[t] = e;
  atomicAdd(&segsum[s], e);
}

__global__ __launch_bounds__(256) void pool_kernel(
    const float* __restrict__ ps, const float* __restrict__ wtmp,
    const float* __restrict__ segsum, const unsigned* __restrict__ seg_lo,
    const unsigned* __restrict__ seg_hi, float* __restrict__ pooled) {
  __shared__ float wch[256];
  const int b = blockIdx.x, tid = threadIdx.x;
  const unsigned lo = seg_lo[b], hi = seg_hi[b];
  float acc0 = 0.0f, acc1 = 0.0f;
  if (lo < hi) {
    const float inv = 1.0f / segsum[b];
    for (unsigned t0 = lo; t0 < hi; t0 += 256) {
      const unsigned n = (hi - t0 < 256u) ? (hi - t0) : 256u;
      if (tid < (int)n) wch[tid] = wtmp[t0 + tid] * inv;
      __syncthreads();
      for (unsigned i = 0; i < n; ++i) {
        const float w = wch[i];
        const float* pr = ps + (size_t)(t0 + i) * XDIM;
        acc0 += w * pr[tid];
        acc1 += w * pr[tid + 256];
      }
      __syncthreads();
    }
  }
  pooled[(size_t)b * XDIM + tid] = acc0;
  pooled[(size_t)b * XDIM + tid + 256] = acc1;
}

// ---------------- final FC: (64,512)->relu(512)->2 -------------------------
__global__ __launch_bounds__(256) void final_fc_kernel(
    const float* __restrict__ pooled, const float* __restrict__ Wf1,
    const float* __restrict__ bf1, const float* __restrict__ Wf2,
    const float* __restrict__ bf2, float* __restrict__ out) {
  __shared__ float pl[HDIM];
  __shared__ float red[2];
  const int b = blockIdx.x, tid = threadIdx.x;
  pl[tid] = pooled[(size_t)b * XDIM + tid];
  pl[tid + 256] = pooled[(size_t)b * XDIM + tid + 256];
  if (tid < 2) red[tid] = 0.0f;
  __syncthreads();
  float p0 = 0.0f, p1 = 0.0f;
  #pragma unroll
  for (int jj = 0; jj < 2; ++jj) {
    const int j = tid + jj * 256;
    float h = bf1[j];
    for (int x = 0; x < HDIM; ++x) h += pl[x] * Wf1[(size_t)x * HDIM + j];
    h = h > 0.0f ? h : 0.0f;
    p0 += h * Wf2[j * 2 + 0];
    p1 += h * Wf2[j * 2 + 1];
  }
  atomicAdd(&red[0], p0);
  atomicAdd(&red[1], p1);
  __syncthreads();
  if (tid < 2) out[b * 2 + tid] = red[tid] + bf2[tid];
}

// ---------------- launch ---------------------------------------------------
extern "C" void kernel_launch(void* const* d_in, const int* in_sizes, int n_in,
                              void* d_out, int out_size, void* d_ws, size_t ws_size,
                              hipStream_t stream) {
  (void)in_sizes; (void)n_in; (void)out_size; (void)ws_size;
  const float* ps  = (const float*)d_in[0];
  const float* io  = (const float*)d_in[1];
  const int*   seg = (const int*)d_in[2];
  const float* W1  = (const float*)d_in[3];
  const float* b1  = (const float*)d_in[4];
  const float* W2  = (const float*)d_in[5];
  const float* b2  = (const float*)d_in[6];
  const float* W3  = (const float*)d_in[7];
  const float* b3  = (const float*)d_in[8];
  const float* Wf1 = (const float*)d_in[9];
  const float* bf1 = (const float*)d_in[10];
  const float* Wf2 = (const float*)d_in[11];
  const float* bf2 = (const float*)d_in[12];
  float* out = (float*)d_out;

  char* base = (char*)d_ws;
  size_t off = 0;
  auto alloc = [&](size_t bytes) -> char* {
    char* p = base + off;
    off = (off + bytes + 255) & ~(size_t)255;
    return p;
  };
  bf16* w1aT       = (bf16*)alloc((size_t)XDIM * HDIM * 2);   // (n,k) 512x512
  bf16* w2T        = (bf16*)alloc((size_t)HDIM * HDIM * 2);   // (n,k) 512x512
  bf16* w1bT       = (bf16*)alloc((size_t)HDIM * YDIM * 2);   // (n,y) 512x2560
  bf16* ioB        = (bf16*)alloc((size_t)BSEG * YDIM * 2);   // 64x2560
  float* seg_bias  = (float*)alloc((size_t)BSEG * HDIM * 4);  // 64x512
  float* logits    = (float*)alloc((size_t)T_TOK * 4);
  float* wtmp      = (float*)alloc((size_t)T_TOK * 4);
  unsigned* segmax_key = (unsigned*)alloc(BSEG * 4);
  float* segsum    = (float*)alloc(BSEG * 4);
  unsigned* seg_lo = (unsigned*)alloc(BSEG * 4);
  unsigned* seg_hi = (unsigned*)alloc(BSEG * 4);
  float* pooled    = (float*)alloc((size_t)BSEG * XDIM * 4);

  init_kernel<<<(BSEG * XDIM + 255) / 256, 256, 0, stream>>>(pooled, segmax_key, segsum,
                                                             seg_lo, seg_hi);
  conv_kernel<<<(BSEG * YDIM + 255) / 256, 256, 0, stream>>>(io, ioB, BSEG * YDIM);
  tconv_kernel<<<(XDIM * HDIM + 255) / 256, 256, 0, stream>>>(W1, w1aT, XDIM, HDIM);
  tconv_kernel<<<(YDIM * HDIM + 255) / 256, 256, 0, stream>>>(W1 + (size_t)XDIM * HDIM,
                                                              w1bT, YDIM, HDIM);
  tconv_kernel<<<(HDIM * HDIM + 255) / 256, 256, 0, stream>>>(W2, w2T, HDIM, HDIM);

  seg_bias_kernel<<<16, 256, 0, stream>>>(ioB, w1bT, b1, seg_bias);
  mlp_logits_kernel<<<T_TOK / 64, 256, 0, stream>>>(ps, seg, w1aT, w2T, seg_bias,
                                                    b2, W3, b3, logits);
  seg_stats_kernel<<<T_TOK / 256, 256, 0, stream>>>(logits, seg, segmax_key, seg_lo, seg_hi);
  exp_sum_kernel<<<T_TOK / 256, 256, 0, stream>>>(logits, seg, segmax_key, wtmp, segsum);
  pool_kernel<<<BSEG, 256, 0, stream>>>(ps, wtmp, segsum, seg_lo, seg_hi, pooled);
  final_fc_kernel<<<BSEG, 256, 0, stream>>>(pooled, Wf1, bf1, Wf2, bf2, out);
}
